// DeepSeekSparseAttention_87857851007603
// MI455X (gfx1250) — compile-verified, matrix operand order checked by reading, not run on hardware
//
#include <hip/hip_runtime.h>
#include <hip/hip_bf16.h>
#include <math.h>

#define D_MODEL 768
#define N_HEADS 12
#define D_HEAD  64
#define SEQ     2048
#define BATCH   2
#define HALF_WIN 64
#define QK_SCALE 0.125f  // 1/sqrt(64)

typedef __bf16 bf16_t;
typedef __attribute__((ext_vector_type(8)))  bf16_t bf16x8;
typedef __attribute__((ext_vector_type(16))) bf16_t bf16x16;
typedef __attribute__((ext_vector_type(8)))  float  f32x8;

static __device__ __forceinline__ bf16x16 make_frag(bf16x8 lo, bf16x8 hi) {
  return __builtin_shufflevector(lo, hi, 0,1,2,3,4,5,6,7,8,9,10,11,12,13,14,15);
}

// 16x32 bf16 tile fragment, row-major storage.
// ISA layout: lanes 0-15 row M=lane, K {0..7}U{16..23}; lanes 16-31 row M=lane-16, K {8..15}U{24..31}.
// => per lane: two contiguous 16B loads.  Same pattern serves B when B^T is stored row-major.
static __device__ __forceinline__ bf16x16 load_frag(const bf16_t* base, int stride, int lane) {
  int m  = lane & 15;
  int kb = (lane >> 4) << 3;
  const bf16_t* p = base + (size_t)m * stride + kb;
  return make_frag(*(const bf16x8*)(p), *(const bf16x8*)(p + 16));
}

static __device__ __forceinline__ int iclamp(int v, int lo, int hi) {
  return v < lo ? lo : (v > hi ? hi : v);
}

// Rows row0+(lane&15) clamped to [0,rowMax]; column pattern shifted by kofs.
static __device__ __forceinline__ bf16x16 load_frag_rowclamp(const bf16_t* base, int stride, int lane,
                                                             int row0, int rowMax, int kofs) {
  int r  = iclamp(row0 + (lane & 15), 0, rowMax);
  int kb = ((lane >> 4) << 3) + kofs;
  const bf16_t* p = base + (size_t)r * stride + kb;
  return make_frag(*(const bf16x8*)(p), *(const bf16x8*)(p + 16));
}

// Rows row0+(lane&15); column windows clamped (multiples of 8 stay aligned).
static __device__ __forceinline__ bf16x16 load_frag_colclamp(const bf16_t* base, int stride, int lane,
                                                             int row0, int col0, int colHi) {
  int r  = row0 + (lane & 15);
  int c0 = iclamp(col0 + ((lane >> 4) << 3), 0, colHi);
  int c1 = iclamp(col0 + ((lane >> 4) << 3) + 16, 0, colHi);
  const bf16_t* p = base + (size_t)r * stride;
  return make_frag(*(const bf16x8*)(p + c0), *(const bf16x8*)(p + c1));
}

static __device__ __forceinline__ f32x8 wmma_bf16(bf16x16 a, bf16x16 b, f32x8 c) {
  return __builtin_amdgcn_wmma_f32_16x16x32_bf16(false, a, false, b, (short)0, c, false, false);
}

// ---------------- pack kernels ----------------
__global__ void pack_x(const float* __restrict__ x, bf16_t* __restrict__ xb, int n) {
  int i = blockIdx.x * blockDim.x + threadIdx.x;
  if (i < n) xb[i] = (bf16_t)x[i];
}

// Fused Wqkv^T [2304][768] bf16: row n = column n' of Wq/Wk/Wv
__global__ void pack_wqkv(const float* __restrict__ Wq, const float* __restrict__ Wk,
                          const float* __restrict__ Wv, bf16_t* __restrict__ wT) {
  int i = blockIdx.x * blockDim.x + threadIdx.x;
  if (i >= 3 * D_MODEL * D_MODEL) return;
  int n = i / D_MODEL, k = i % D_MODEL;
  float v;
  if (n < D_MODEL)            v = Wq[(size_t)k * D_MODEL + n];
  else if (n < 2 * D_MODEL)   v = Wk[(size_t)k * D_MODEL + (n - D_MODEL)];
  else                        v = Wv[(size_t)k * D_MODEL + (n - 2 * D_MODEL)];
  wT[i] = (bf16_t)v;
}

__global__ void pack_wo(const float* __restrict__ Wo, bf16_t* __restrict__ wT) {
  int i = blockIdx.x * blockDim.x + threadIdx.x;
  if (i >= D_MODEL * D_MODEL) return;
  int n = i / D_MODEL, k = i % D_MODEL;
  wT[i] = (bf16_t)Wo[(size_t)k * D_MODEL + n];
}

// ---------------- fused QKV projection GEMM ----------------
// [4096 x 768] @ [768 x 2304]; wave tile 16 rows x 64 cols; block = 8 waves = 32x256.
__global__ __launch_bounds__(256) void qkv_gemm(const bf16_t* __restrict__ xb,
                                                const bf16_t* __restrict__ wT,
                                                const float* __restrict__ bq,
                                                const float* __restrict__ bk,
                                                const float* __restrict__ bv,
                                                bf16_t* __restrict__ qbuf,
                                                bf16_t* __restrict__ kbuf,
                                                bf16_t* __restrict__ vT) {
  const int wid  = threadIdx.x >> 5;
  const int lane = threadIdx.x & 31;
  const int m0 = (blockIdx.x * 2 + (wid >> 2)) * 16;
  const int n0 = blockIdx.y * 256 + (wid & 3) * 64;
  const bf16_t* abase = xb + (size_t)m0 * D_MODEL;
  f32x8 acc0 = {}, acc1 = {}, acc2 = {}, acc3 = {};
  for (int k = 0; k < D_MODEL; k += 32) {
    bf16x16 a = load_frag(abase + k, D_MODEL, lane);
    acc0 = wmma_bf16(a, load_frag(wT + (size_t)(n0 +  0) * D_MODEL + k, D_MODEL, lane), acc0);
    acc1 = wmma_bf16(a, load_frag(wT + (size_t)(n0 + 16) * D_MODEL + k, D_MODEL, lane), acc1);
    acc2 = wmma_bf16(a, load_frag(wT + (size_t)(n0 + 32) * D_MODEL + k, D_MODEL, lane), acc2);
    acc3 = wmma_bf16(a, load_frag(wT + (size_t)(n0 + 48) * D_MODEL + k, D_MODEL, lane), acc3);
  }
  const int col   = lane & 15;
  const int rbase = (lane >> 4) * 8;
  f32x8 accs[4] = {acc0, acc1, acc2, acc3};
  #pragma unroll
  for (int nt = 0; nt < 4; ++nt) {
    int n = n0 + nt * 16 + col;
    #pragma unroll
    for (int r = 0; r < 8; ++r) {
      int m  = m0 + rbase + r;
      int bb = m >> 11;          // batch
      int t  = m & (SEQ - 1);    // token
      float v = accs[nt][r];
      if (n < D_MODEL) {
        int hh = n >> 6, dd = n & 63;
        qbuf[(((size_t)bb * N_HEADS + hh) * SEQ + t) * D_HEAD + dd] = (bf16_t)((v + bq[n]) * QK_SCALE);
      } else if (n < 2 * D_MODEL) {
        int n2 = n - D_MODEL; int hh = n2 >> 6, dd = n2 & 63;
        kbuf[(((size_t)bb * N_HEADS + hh) * SEQ + t) * D_HEAD + dd] = (bf16_t)(v + bk[n2]);
      } else {
        int n2 = n - 2 * D_MODEL; int hh = n2 >> 6, dd = n2 & 63;
        vT[(((size_t)bb * N_HEADS + hh) * D_HEAD + dd) * SEQ + t] = (bf16_t)(v + bv[n2]);
      }
    }
  }
}

// ---------------- flash-style local attention ----------------
// One wave per 16-query tile; stream 5 pairs of 16-key tiles covering [t0-64, t0+95].
__global__ __launch_bounds__(256) void attn_local(const bf16_t* __restrict__ qbuf,
                                                  const bf16_t* __restrict__ kbuf,
                                                  const bf16_t* __restrict__ vT,
                                                  bf16_t* __restrict__ ctx) {
  __shared__ __align__(16) bf16_t pstage[8 * 16 * 32];
  const int wid  = threadIdx.x >> 5;
  const int lane = threadIdx.x & 31;
  const int gw   = blockIdx.x * 8 + wid;   // 0..3071
  const int qt   = gw & 127;
  const int bh   = gw >> 7;                // b*12+h
  const int t0   = qt << 4;
  const int b    = bh / N_HEADS;
  const int h    = bh % N_HEADS;

  bf16_t* plds = pstage + wid * (16 * 32);

  const bf16_t* qb = qbuf + ((size_t)bh * SEQ + t0) * D_HEAD;
  const bf16_t* kb = kbuf + (size_t)bh * SEQ * D_HEAD;
  const bf16_t* vb = vT   + (size_t)bh * D_HEAD * SEQ;

  const bf16x16 aq0 = load_frag(qb,      D_HEAD, lane);  // k 0..31
  const bf16x16 aq1 = load_frag(qb + 32, D_HEAD, lane);  // k 32..63

  f32x8 o0 = {}, o1 = {}, o2 = {}, o3 = {};
  float mrow[8], srow[8];
  #pragma unroll
  for (int r = 0; r < 8; ++r) { mrow[r] = -__builtin_inff(); srow[r] = 0.0f; }

  const int col   = lane & 15;
  const int rbase = (lane >> 4) * 8;

  for (int p = 0; p < 5; ++p) {
    const int ks = t0 - HALF_WIN + p * 32;
    f32x8 sfrag[2];
    #pragma unroll
    for (int tt = 0; tt < 2; ++tt) {
      const int kt = ks + tt * 16;
      bf16x16 bk0 = load_frag_rowclamp(kb, D_HEAD, lane, kt, SEQ - 1, 0);
      bf16x16 bk1 = load_frag_rowclamp(kb, D_HEAD, lane, kt, SEQ - 1, 32);
      f32x8 s = {};
      s = wmma_bf16(aq0, bk0, s);
      s = wmma_bf16(aq1, bk1, s);
      #pragma unroll
      for (int r = 0; r < 8; ++r) {
        int qi = t0 + rbase + r;
        int ki = kt + col;
        int dd = qi - ki;
        bool valid = (ki >= 0) && (ki < SEQ) && (dd <= HALF_WIN) && (dd >= -HALF_WIN);
        s[r] = valid ? s[r] : -__builtin_inff();
      }
      sfrag[tt] = s;
    }
    // online softmax update; row reductions across the 16-lane half-groups
    float nm[8], corr[8];
    #pragma unroll
    for (int r = 0; r < 8; ++r) {
      float t = fmaxf(sfrag[0][r], sfrag[1][r]);
      t = fmaxf(t, __shfl_xor(t, 1, 32));
      t = fmaxf(t, __shfl_xor(t, 2, 32));
      t = fmaxf(t, __shfl_xor(t, 4, 32));
      t = fmaxf(t, __shfl_xor(t, 8, 32));
      float newm = fmaxf(mrow[r], t);
      corr[r] = (mrow[r] == -__builtin_inff()) ? 0.0f : __expf(mrow[r] - newm);
      nm[r] = newm;
    }
    #pragma unroll
    for (int r = 0; r < 8; ++r) {
      o0[r] *= corr[r]; o1[r] *= corr[r]; o2[r] *= corr[r]; o3[r] *= corr[r];
    }
    #pragma unroll
    for (int r = 0; r < 8; ++r) {
      float p0 = (sfrag[0][r] == -__builtin_inff()) ? 0.0f : __expf(sfrag[0][r] - nm[r]);
      float p1 = (sfrag[1][r] == -__builtin_inff()) ? 0.0f : __expf(sfrag[1][r] - nm[r]);
      plds[(rbase + r) * 32 + col]      = (bf16_t)p0;   // stage P (C-layout -> A-layout via LDS)
      plds[(rbase + r) * 32 + 16 + col] = (bf16_t)p1;
      float rs = p0 + p1;
      rs += __shfl_xor(rs, 1, 32);
      rs += __shfl_xor(rs, 2, 32);
      rs += __shfl_xor(rs, 4, 32);
      rs += __shfl_xor(rs, 8, 32);
      srow[r] = srow[r] * corr[r] + rs;
      mrow[r] = nm[r];
    }
    asm volatile("s_wait_dscnt 0" ::: "memory");  // wave-local LDS store->load ordering
    bf16x16 ap = load_frag(plds, 32, lane);       // P as 16x32 A fragment
    o0 = wmma_bf16(ap, load_frag_colclamp(vb, SEQ, lane,  0, ks, SEQ - 8), o0);
    o1 = wmma_bf16(ap, load_frag_colclamp(vb, SEQ, lane, 16, ks, SEQ - 8), o1);
    o2 = wmma_bf16(ap, load_frag_colclamp(vb, SEQ, lane, 32, ks, SEQ - 8), o2);
    o3 = wmma_bf16(ap, load_frag_colclamp(vb, SEQ, lane, 48, ks, SEQ - 8), o3);
  }
  #pragma unroll
  for (int r = 0; r < 8; ++r) {
    float inv = 1.0f / srow[r];
    int tok = b * SEQ + t0 + rbase + r;
    bf16_t* crow = ctx + (size_t)tok * D_MODEL + h * D_HEAD;
    crow[ 0 + col] = (bf16_t)(o0[r] * inv);
    crow[16 + col] = (bf16_t)(o1[r] * inv);
    crow[32 + col] = (bf16_t)(o2[r] * inv);
    crow[48 + col] = (bf16_t)(o3[r] * inv);
  }
}

// ---------------- output projection GEMM ----------------
__global__ __launch_bounds__(256) void out_gemm(const bf16_t* __restrict__ ctx,
                                                const bf16_t* __restrict__ woT,
                                                const float* __restrict__ bo,
                                                float* __restrict__ out) {
  const int wid  = threadIdx.x >> 5;
  const int lane = threadIdx.x & 31;
  const int m0 = (blockIdx.x * 2 + (wid >> 2)) * 16;
  const int n0 = blockIdx.y * 256 + (wid & 3) * 64;
  const bf16_t* abase = ctx + (size_t)m0 * D_MODEL;
  f32x8 acc0 = {}, acc1 = {}, acc2 = {}, acc3 = {};
  for (int k = 0; k < D_MODEL; k += 32) {
    bf16x16 a = load_frag(abase + k, D_MODEL, lane);
    acc0 = wmma_bf16(a, load_frag(woT + (size_t)(n0 +  0) * D_MODEL + k, D_MODEL, lane), acc0);
    acc1 = wmma_bf16(a, load_frag(woT + (size_t)(n0 + 16) * D_MODEL + k, D_MODEL, lane), acc1);
    acc2 = wmma_bf16(a, load_frag(woT + (size_t)(n0 + 32) * D_MODEL + k, D_MODEL, lane), acc2);
    acc3 = wmma_bf16(a, load_frag(woT + (size_t)(n0 + 48) * D_MODEL + k, D_MODEL, lane), acc3);
  }
  const int col   = lane & 15;
  const int rbase = (lane >> 4) * 8;
  f32x8 accs[4] = {acc0, acc1, acc2, acc3};
  #pragma unroll
  for (int nt = 0; nt < 4; ++nt) {
    int n = n0 + nt * 16 + col;
    #pragma unroll
    for (int r = 0; r < 8; ++r) {
      int m = m0 + rbase + r;
      out[(size_t)m * D_MODEL + n] = accs[nt][r] + bo[n];
    }
  }
}

extern "C" void kernel_launch(void* const* d_in, const int* in_sizes, int n_in,
                              void* d_out, int out_size, void* d_ws, size_t ws_size,
                              hipStream_t stream) {
  (void)in_sizes; (void)n_in; (void)out_size; (void)ws_size;
  const float* x  = (const float*)d_in[0];
  const float* Wq = (const float*)d_in[1];
  const float* bq = (const float*)d_in[2];
  const float* Wk = (const float*)d_in[3];
  const float* bk = (const float*)d_in[4];
  const float* Wv = (const float*)d_in[5];
  const float* bv = (const float*)d_in[6];
  const float* Wo = (const float*)d_in[7];
  const float* bo = (const float*)d_in[8];
  float* out = (float*)d_out;

  char* ws = (char*)d_ws;
  size_t off = 0;
  auto take = [&](size_t bytes) -> char* {
    off = (off + 255) & ~(size_t)255;
    char* p = ws + off;
    off += bytes;
    return p;
  };
  const size_t MTOK = (size_t)BATCH * SEQ;               // 4096
  bf16_t* xb    = (bf16_t*)take(MTOK * D_MODEL * 2);
  bf16_t* wqkvT = (bf16_t*)take((size_t)3 * D_MODEL * D_MODEL * 2);
  bf16_t* woT   = (bf16_t*)take((size_t)D_MODEL * D_MODEL * 2);
  bf16_t* qbuf  = (bf16_t*)take(MTOK * D_MODEL * 2);
  bf16_t* kbuf  = (bf16_t*)take(MTOK * D_MODEL * 2);
  bf16_t* vTb   = (bf16_t*)take(MTOK * D_MODEL * 2);
  bf16_t* ctx   = (bf16_t*)take(MTOK * D_MODEL * 2);

  const int nx = (int)(MTOK * D_MODEL);
  pack_x<<<(nx + 255) / 256, 256, 0, stream>>>(x, xb, nx);
  pack_wqkv<<<(3 * D_MODEL * D_MODEL + 255) / 256, 256, 0, stream>>>(Wq, Wk, Wv, wqkvT);
  pack_wo<<<(D_MODEL * D_MODEL + 255) / 256, 256, 0, stream>>>(Wo, woT);
  qkv_gemm<<<dim3(128, 9), 256, 0, stream>>>(xb, wqkvT, bq, bk, bv, qbuf, kbuf, vTb);
  attn_local<<<384, 256, 0, stream>>>(qbuf, kbuf, vTb, ctx);
  out_gemm<<<dim3(128, 3), 256, 0, stream>>>(ctx, woT, bo, out);
}
